// ScoringNet_51427938402920
// MI455X (gfx1250) — matrix operand order repaired and checked by reading, not compiled, not run on hardware
//
#include <hip/hip_runtime.h>

// ---------------------------------------------------------------------------
// CDNA5 (gfx1250) implementation of ScoringNet.
// All GEMM-shaped work runs through v_wmma_f32_16x16x32_bf16 (f32 accumulate,
// bf16 operands produced on the LDS staging path). wave32 / WGP aware:
// 256-thread blocks = 8 waves, each wave owns a 32x64 patch of a 128x128 tile.
// FULL-path: float4 global loads, b64 LDS stores, double-buffered LDS tiles,
// branch-free pipelined K loop (last iteration peeled), all 6 fragments
// loaded before an uninterrupted 8-WMMA burst.
// ---------------------------------------------------------------------------

typedef __attribute__((ext_vector_type(16))) __bf16 v16bf;
typedef __attribute__((ext_vector_type(2)))  __bf16 v2bf;
typedef __attribute__((ext_vector_type(8)))  float  v8f;

#define BM 128
#define BN 128
#define BK 32
#define LDSH 40                 // halves per LDS tile row (BK + 8 pad)
#define TILE_HALVES (BM * LDSH) // one LDS tile in halves

union Frag { unsigned int u[8]; v16bf v; };

__device__ __forceinline__ unsigned short f2bf(float f) {
  unsigned int u = __float_as_uint(f);
  u += 0x7FFFu + ((u >> 16) & 1u);  // round-to-nearest-even
  return (unsigned short)(u >> 16);
}
// pack two f32 -> packed bf16x2 (one v_cvt_pk_bf16_f32 when available)
__device__ __forceinline__ unsigned int pk2(float lo, float hi) {
#if __has_builtin(__builtin_amdgcn_cvt_pk_bf16_f32)
  v2bf r = __builtin_amdgcn_cvt_pk_bf16_f32(lo, hi);
  return __builtin_bit_cast(unsigned int, r);
#else
  unsigned int ul = __float_as_uint(lo);
  unsigned int uh = __float_as_uint(hi);
  ul += 0x7FFFu + ((ul >> 16) & 1u);
  uh += 0x7FFFu + ((uh >> 16) & 1u);
  return __builtin_amdgcn_perm(uh, ul, 0x07060302u);  // {uh[31:16], ul[31:16]}
#endif
}

// C[M,N] = act( scale * (A @ opB) + bias ),  A:[M,K] row-major (lda)
// BT=true : B:[N,K] row-major (ldb)  -> C = A * B^T   ("NT", weights / Q.K^T)
// BT=false: B:[K,N] row-major (ldb)  -> C = A * B     ("NN", P.V)
// FULL=true: M%128==0, N%128==0, K%32==0, lda/ldb%4==0, 16B-aligned bases.
// Batched over blockIdx.z; z = (zo,zi), zi in [0,IC): ptr += zo*s?o + zi*s?i.
template <int ACT, bool BT, bool FULL>
__global__ void __launch_bounds__(256)
gemm_kernel(const float* __restrict__ A, int lda, long long sAo, long long sAi,
            const float* __restrict__ B, int ldb, long long sBo, long long sBi,
            float* __restrict__ C, int ldc, long long sCo, long long sCi,
            const float* __restrict__ bias,
            int M, int N, int K, int IC, float scale)
{
  __shared__ unsigned short As[2 * TILE_HALVES];
  __shared__ unsigned short Bs[2 * TILE_HALVES];

  const int z  = blockIdx.z;
  const int zo = z / IC;
  const int zi = z - zo * IC;
  A += zo * sAo + zi * sAi;
  B += zo * sBo + zi * sBi;
  C += zo * sCo + zi * sCi;

  const int m0   = blockIdx.y * BM;
  const int n0   = blockIdx.x * BN;
  const int tid  = threadIdx.x;
  const int lane = tid & 31;
  const int wave = tid >> 5;
  const int wm   = wave >> 1;   // 0..3 : 32-row strip
  const int wn   = wave & 1;    // 0..1 : 64-col strip
  const int half = lane >> 4;   // lane half (CDNA5 WMMA layout)
  const int lr   = lane & 15;

  v8f acc[2][4];
#pragma unroll
  for (int i = 0; i < 2; ++i)
#pragma unroll
    for (int j = 0; j < 4; ++j)
#pragma unroll
      for (int r = 0; r < 8; ++r) acc[i][j][r] = 0.0f;

  const int nk = (K + BK - 1) / BK;

  // load ALL fragments (2 A + 4 B = 12 ds_load_b128), then an uninterrupted
  // burst of 8 WMMAs -- avoids dscnt stalls from register recycling
  auto compute = [&](int off) {
    Frag af[2];
    Frag bf[4];
    const int kb2a = half * 4;  // A kbase 0/8 -> u32 0/4
#pragma unroll
    for (int mt = 0; mt < 2; ++mt) {
      const unsigned int* rp =
          (const unsigned int*)&As[off + (wm * 32 + mt * 16 + lr) * LDSH];
#pragma unroll
      for (int j = 0; j < 8; ++j) af[mt].u[j] = rp[kb2a + (j < 4 ? j : j + 4)];
    }
    const int kb2b = half * 4;  // B kbase 0/8 -> u32 0/4
#pragma unroll
    for (int nt = 0; nt < 4; ++nt) {
      const unsigned int* rp =
          (const unsigned int*)&Bs[off + (wn * 64 + nt * 16 + lr) * LDSH];
#pragma unroll
      for (int j = 0; j < 8; ++j) bf[nt].u[j] = rp[kb2b + (j < 4 ? j : j + 4)];
    }
#pragma unroll
    for (int nt = 0; nt < 4; ++nt)
#pragma unroll
      for (int mt = 0; mt < 2; ++mt)
        acc[mt][nt] = __builtin_amdgcn_wmma_f32_16x16x32_bf16(
            false, af[mt].v, false, bf[nt].v, (short)0, acc[mt][nt], false, false);
  };

  if constexpr (FULL) {
    float4 pa[4], pb[4];
    auto loadTiles = [&](int kt) {
      const int k0 = kt * BK;
#pragma unroll
      for (int i = 0; i < 4; ++i) {
        int f = i * 256 + tid;
        int r = f >> 3, c4 = (f & 7) * 4;
        pa[i] = *(const float4*)(A + (long long)(m0 + r) * lda + k0 + c4);
      }
      if constexpr (BT) {
#pragma unroll
        for (int i = 0; i < 4; ++i) {
          int f = i * 256 + tid;
          int r = f >> 3, c4 = (f & 7) * 4;
          pb[i] = *(const float4*)(B + (long long)(n0 + r) * ldb + k0 + c4);
        }
      } else {
#pragma unroll
        for (int i = 0; i < 4; ++i) {
          int f = i * 256 + tid;
          int kk = f >> 5, n4 = (f & 31) * 4;
          pb[i] = *(const float4*)(B + (long long)(k0 + kk) * ldb + n0 + n4);
        }
      }
    };
    auto storeTiles = [&](int idx) {
      const int off = idx * TILE_HALVES;
#pragma unroll
      for (int i = 0; i < 4; ++i) {
        int f = i * 256 + tid;
        int r = f >> 3, c4 = (f & 7) * 4;
        uint2 p;
        p.x = pk2(pa[i].x, pa[i].y);
        p.y = pk2(pa[i].z, pa[i].w);
        *(uint2*)&As[off + r * LDSH + c4] = p;
      }
      if constexpr (BT) {
#pragma unroll
        for (int i = 0; i < 4; ++i) {
          int f = i * 256 + tid;
          int r = f >> 3, c4 = (f & 7) * 4;
          uint2 p;
          p.x = pk2(pb[i].x, pb[i].y);
          p.y = pk2(pb[i].z, pb[i].w);
          *(uint2*)&Bs[off + r * LDSH + c4] = p;
        }
      } else {
#pragma unroll
        for (int i = 0; i < 4; ++i) {
          int f = i * 256 + tid;
          int kk = f >> 5, n4 = (f & 31) * 4;
          unsigned int pxy = pk2(pb[i].x, pb[i].y);
          unsigned int pzw = pk2(pb[i].z, pb[i].w);
          Bs[off + (n4 + 0) * LDSH + kk] = (unsigned short)pxy;
          Bs[off + (n4 + 1) * LDSH + kk] = (unsigned short)(pxy >> 16);
          Bs[off + (n4 + 2) * LDSH + kk] = (unsigned short)pzw;
          Bs[off + (n4 + 3) * LDSH + kk] = (unsigned short)(pzw >> 16);
        }
      }
    };

    loadTiles(0);
    storeTiles(0);
    int kt = 0;
    for (; kt < nk - 1; ++kt) {          // branch-free steady state
      __syncthreads();
      loadTiles(kt + 1);                 // prefetch overlaps WMMAs below
      compute((kt & 1) * TILE_HALVES);
      storeTiles((kt + 1) & 1);          // other buffer: free since kt-1 sync
    }
    __syncthreads();
    compute((kt & 1) * TILE_HALVES);     // peeled last iteration
  } else {
    // generic path (small encoder GEMMs): clamped loads, no exec branching
    for (int kt = 0; kt < nk; ++kt) {
      const int k0 = kt * BK;
#pragma unroll
      for (int i = 0; i < 16; ++i) {
        int e = i * 256 + tid;
        int r = e >> 5, cc = e & 31;
        int gm = m0 + r, gk = k0 + cc;
        int cm = gm < M ? gm : M - 1;
        int ck = gk < K ? gk : K - 1;
        float v = A[(long long)cm * lda + ck];
        As[r * LDSH + cc] = f2bf((gm < M && gk < K) ? v : 0.0f);
      }
      if constexpr (BT) {
#pragma unroll
        for (int i = 0; i < 16; ++i) {
          int e = i * 256 + tid;
          int r = e >> 5, cc = e & 31;
          int gn = n0 + r, gk = k0 + cc;
          int cn = gn < N ? gn : N - 1;
          int ck = gk < K ? gk : K - 1;
          float v = B[(long long)cn * ldb + ck];
          Bs[r * LDSH + cc] = f2bf((gn < N && gk < K) ? v : 0.0f);
        }
      } else {
#pragma unroll
        for (int i = 0; i < 16; ++i) {
          int e = i * 256 + tid;
          int kk = e >> 7, n = e & 127;
          int gk = k0 + kk, gn = n0 + n;
          int ck = gk < K ? gk : K - 1;
          int cn = gn < N ? gn : N - 1;
          float v = B[(long long)ck * ldb + cn];
          Bs[n * LDSH + kk] = f2bf((gk < K && gn < N) ? v : 0.0f);
        }
      }
      __syncthreads();
      compute(0);
      __syncthreads();
    }
  }

  // ---- epilogue: C/D layout VGPR r -> M=r (lanes 0-15) / r+8 (lanes 16-31)
#pragma unroll
  for (int mt = 0; mt < 2; ++mt)
#pragma unroll
    for (int nt = 0; nt < 4; ++nt) {
      int n = n0 + wn * 64 + nt * 16 + lr;
      if (!FULL && n >= N) continue;
      float bv = bias ? bias[n] : 0.0f;
#pragma unroll
      for (int r = 0; r < 8; ++r) {
        int m = m0 + wm * 32 + mt * 16 + half * 8 + r;
        if (!FULL && m >= M) continue;
        float v = acc[mt][nt][r] * scale + bv;
        if (ACT == 1) v = fmaxf(v, 0.0f);
        if (ACT == 2) v = (v > 0.0f) ? v : 0.01f * v;
        C[(long long)m * ldc + n] = v;
      }
    }
}

// one block per (b,h,q) row of 512 keys; mask keys >= len with -1e9 (NEG)
__global__ void __launch_bounds__(256)
softmax_kernel(float* __restrict__ s, const int* __restrict__ lens)
{
  __shared__ float red[256];
  const int row = blockIdx.x;          // (b*2+h)*512 + q
  const int b   = row >> 10;           // H*Tq = 1024
  const int len = lens[b];
  float* sr = s + (long long)row * 512;
  const int t = threadIdx.x;
  float v0 = (t < len) ? sr[t] : -1e9f;
  float v1 = (t + 256 < len) ? sr[t + 256] : -1e9f;
  red[t] = fmaxf(v0, v1);
  __syncthreads();
  for (int o = 128; o > 0; o >>= 1) {
    if (t < o) red[t] = fmaxf(red[t], red[t + o]);
    __syncthreads();
  }
  float mx = red[0];
  __syncthreads();
  float e0 = __expf(v0 - mx), e1 = __expf(v1 - mx);
  red[t] = e0 + e1;
  __syncthreads();
  for (int o = 128; o > 0; o >>= 1) {
    if (t < o) red[t] += red[t + o];
    __syncthreads();
  }
  float inv = 1.0f / red[0];
  sr[t] = e0 * inv;
  sr[t + 256] = e1 * inv;
}

// x = LN(x + y) * g + b   -- one wave32 per 256-wide row, 8 rows per block
__global__ void __launch_bounds__(256)
add_ln_kernel(float* __restrict__ x, const float* __restrict__ y,
              const float* __restrict__ g, const float* __restrict__ b, int M)
{
  const int lane = threadIdx.x & 31;
  const int row  = blockIdx.x * 8 + (threadIdx.x >> 5);
  if (row >= M) return;
  float* xr = x + (long long)row * 256;
  const float* yr = y + (long long)row * 256;
  float v[8];
  float s = 0.0f;
#pragma unroll
  for (int j = 0; j < 8; ++j) {
    v[j] = xr[lane + 32 * j] + yr[lane + 32 * j];
    s += v[j];
  }
#pragma unroll
  for (int o = 16; o > 0; o >>= 1) s += __shfl_xor(s, o, 32);
  const float mean = s * (1.0f / 256.0f);
  float q = 0.0f;
#pragma unroll
  for (int j = 0; j < 8; ++j) { float d = v[j] - mean; q += d * d; }
#pragma unroll
  for (int o = 16; o > 0; o >>= 1) q += __shfl_xor(q, o, 32);
  const float rstd = rsqrtf(q * (1.0f / 256.0f) + 1e-5f);
#pragma unroll
  for (int j = 0; j < 8; ++j) {
    int col = lane + 32 * j;
    xr[col] = (v[j] - mean) * rstd * g[col] + b[col];
  }
}

// point / number positional encoders: per token emit
// [ per point, per coord: sin(c*sw+sb)[8], cos(c*cw+cb)[8] ] ++ num*w+b [8]
__global__ void __launch_bounds__(256)
encode_kernel(const float* __restrict__ pts, const float* __restrict__ nums,
              const float* __restrict__ sw, const float* __restrict__ sb,
              const float* __restrict__ cw, const float* __restrict__ cb,
              const float* __restrict__ nw, const float* __restrict__ nb,
              float* __restrict__ out, int nTok, int P, int hasNum)
{
  int t = blockIdx.x * 256 + threadIdx.x;
  if (t >= nTok) return;
  const int Kout = P * 32 + (hasNum ? 8 : 0);
  float* o = out + (long long)t * Kout;
  const float* p = pts + (long long)t * P * 2;
  int idx = 0;
  for (int pp = 0; pp < P; ++pp)
    for (int cc = 0; cc < 2; ++cc) {
      float c = p[pp * 2 + cc];
      for (int j = 0; j < 8; ++j) o[idx++] = __sinf(c * sw[j] + sb[j]);
      for (int j = 0; j < 8; ++j) o[idx++] = __cosf(c * cw[j] + cb[j]);
    }
  if (hasNum) {
    float xv = nums[t];
    for (int j = 0; j < 8; ++j) o[idx++] = xv * nw[j] + nb[j];
  }
}

// ---------------------------------------------------------------------------
// host-side orchestration
// ---------------------------------------------------------------------------
static inline bool gemm_full(int M, int N, int K, int lda, int ldb) {
  return (M % BM == 0) && (N % BN == 0) && (K % BK == 0) &&
         (lda % 4 == 0) && (ldb % 4 == 0);
}

static inline void gemm_nt(hipStream_t st, int act,
                           const float* A, int lda, long long sAo, long long sAi,
                           const float* B, int ldb, long long sBo, long long sBi,
                           float* C, int ldc, long long sCo, long long sCi,
                           const float* bias, int M, int N, int K,
                           int batches, int IC, float scale)
{
  dim3 g((N + BN - 1) / BN, (M + BM - 1) / BM, batches), blk(256);
  const bool full = gemm_full(M, N, K, lda, ldb);
#define ARGS A, lda, sAo, sAi, B, ldb, sBo, sBi, C, ldc, sCo, sCi, bias, M, N, K, IC, scale
  if (full) {
    if (act == 0)      gemm_kernel<0, true, true><<<g, blk, 0, st>>>(ARGS);
    else if (act == 1) gemm_kernel<1, true, true><<<g, blk, 0, st>>>(ARGS);
    else               gemm_kernel<2, true, true><<<g, blk, 0, st>>>(ARGS);
  } else {
    if (act == 0)      gemm_kernel<0, true, false><<<g, blk, 0, st>>>(ARGS);
    else if (act == 1) gemm_kernel<1, true, false><<<g, blk, 0, st>>>(ARGS);
    else               gemm_kernel<2, true, false><<<g, blk, 0, st>>>(ARGS);
  }
#undef ARGS
}

static inline void gemm_nn(hipStream_t st,
                           const float* A, int lda, long long sAo, long long sAi,
                           const float* B, int ldb, long long sBo, long long sBi,
                           float* C, int ldc, long long sCo, long long sCi,
                           const float* bias, int M, int N, int K,
                           int batches, int IC, float scale)
{
  dim3 g((N + BN - 1) / BN, (M + BM - 1) / BM, batches), blk(256);
  const bool full = gemm_full(M, N, K, lda, ldb);
#define ARGS A, lda, sAo, sAi, B, ldb, sBo, sBi, C, ldc, sCo, sCi, bias, M, N, K, IC, scale
  if (full) gemm_kernel<0, false, true><<<g, blk, 0, st>>>(ARGS);
  else      gemm_kernel<0, false, false><<<g, blk, 0, st>>>(ARGS);
#undef ARGS
}

// MHA: D=256, H=2, hd=128, B=32, T=512, M = 16384 rows
static void run_mha(hipStream_t st, const float* tgt, const float* mem,
                    const float* w_in, const float* b_in,
                    const float* w_out, const float* b_out,
                    const int* kvlen,
                    float* q, float* k, float* v, float* s, float* attn,
                    float* outp)
{
  const int M = 16384;
  gemm_nt(st, 0, tgt, 256, 0, 0, w_in,          256, 0, 0, q, 256, 0, 0, b_in,       M, 256, 256, 1, 1, 1.0f);
  gemm_nt(st, 0, mem, 256, 0, 0, w_in + 65536,  256, 0, 0, k, 256, 0, 0, b_in + 256, M, 256, 256, 1, 1, 1.0f);
  gemm_nt(st, 0, mem, 256, 0, 0, w_in + 131072, 256, 0, 0, v, 256, 0, 0, b_in + 512, M, 256, 256, 1, 1, 1.0f);
  // S[b,h,q,k] = (Q . K^T) / sqrt(128) -- 64 batches, inner=H selects head slice
  gemm_nt(st, 0, q, 256, 131072, 128, k, 256, 131072, 128,
          s, 512, 524288, 262144, nullptr, 512, 512, 128, 64, 2,
          0.08838834764831845f);
  softmax_kernel<<<65536, 256, 0, st>>>(s, kvlen);
  // O[b,q,h,:] = P . V
  gemm_nn(st, s, 512, 524288, 262144, v, 256, 131072, 128,
          attn, 256, 131072, 128, nullptr, 512, 128, 512, 64, 2, 1.0f);
  gemm_nt(st, 0, attn, 256, 0, 0, w_out, 256, 0, 0, outp, 256, 0, 0, b_out,
          M, 256, 256, 1, 1, 1.0f);
}

struct DecW {
  const float *sa_w, *sa_b, *sa_ow, *sa_ob;
  const float *ca_w, *ca_b, *ca_ow, *ca_ob;
  const float *f1w, *f1b, *f2w, *f2b;
  const float *lng, *lnb;
};

static void run_dec_layer(hipStream_t st, float* tgt, const float* mem,
                          const int* tlen, const int* mlen, int i, const DecW& W,
                          float* q, float* k, float* v, float* s, float* attn,
                          float* proj, float* h)
{
  const long long wIn = (long long)i * 768 * 256;
  const long long wSq = (long long)i * 256 * 256;
  // self-attn + LN
  run_mha(st, tgt, tgt, W.sa_w + wIn, W.sa_b + i * 768, W.sa_ow + wSq,
          W.sa_ob + i * 256, tlen, q, k, v, s, attn, proj);
  add_ln_kernel<<<2048, 256, 0, st>>>(tgt, proj, W.lng + (i * 3 + 0) * 256,
                                      W.lnb + (i * 3 + 0) * 256, 16384);
  // cross-attn + LN
  run_mha(st, tgt, mem, W.ca_w + wIn, W.ca_b + i * 768, W.ca_ow + wSq,
          W.ca_ob + i * 256, mlen, q, k, v, s, attn, proj);
  add_ln_kernel<<<2048, 256, 0, st>>>(tgt, proj, W.lng + (i * 3 + 1) * 256,
                                      W.lnb + (i * 3 + 1) * 256, 16384);
  // FFN (relu) + LN
  gemm_nt(st, 1, tgt, 256, 0, 0, W.f1w + wSq, 256, 0, 0, h, 256, 0, 0,
          W.f1b + i * 256, 16384, 256, 256, 1, 1, 1.0f);
  gemm_nt(st, 0, h, 256, 0, 0, W.f2w + wSq, 256, 0, 0, proj, 256, 0, 0,
          W.f2b + i * 256, 16384, 256, 256, 1, 1, 1.0f);
  add_ln_kernel<<<2048, 256, 0, st>>>(tgt, proj, W.lng + (i * 3 + 2) * 256,
                                      W.lnb + (i * 3 + 2) * 256, 16384);
}

extern "C" void kernel_launch(void* const* d_in, const int* in_sizes, int n_in,
                              void* d_out, int out_size, void* d_ws, size_t ws_size,
                              hipStream_t stream)
{
  (void)in_sizes; (void)n_in; (void)out_size; (void)ws_size;
  const float* o_points   = (const float*)d_in[0];
  const float* o_numbers  = (const float*)d_in[1];
  const float* c_points   = (const float*)d_in[2];
  const float* ar_points  = (const float*)d_in[3];
  const float* ar_numbers = (const float*)d_in[4];
  const int*   o_len      = (const int*)d_in[5];
  const int*   c_len      = (const int*)d_in[6];
  const int*   ar_len     = (const int*)d_in[7];
  const float* pe_sin_w   = (const float*)d_in[8];
  const float* pe_sin_b   = (const float*)d_in[9];
  const float* pe_cos_w   = (const float*)d_in[10];
  const float* pe_cos_b   = (const float*)d_in[11];
  const float* o_num_w    = (const float*)d_in[12];
  const float* o_num_b    = (const float*)d_in[13];
  const float* o_mlp1_w   = (const float*)d_in[14];
  const float* o_mlp1_b   = (const float*)d_in[15];
  const float* o_mlp2_w   = (const float*)d_in[16];
  const float* o_mlp2_b   = (const float*)d_in[17];
  const float* c_mlp1_w   = (const float*)d_in[18];
  const float* c_mlp1_b   = (const float*)d_in[19];
  const float* c_mlp2_w   = (const float*)d_in[20];
  const float* c_mlp2_b   = (const float*)d_in[21];
  const float* ar_num_w   = (const float*)d_in[22];
  const float* ar_num_b   = (const float*)d_in[23];
  const float* ar_mlp1_w  = (const float*)d_in[24];
  const float* ar_mlp1_b  = (const float*)d_in[25];
  const float* ar_mlp2_w  = (const float*)d_in[26];
  const float* ar_mlp2_b  = (const float*)d_in[27];
  DecW W;
  W.sa_w  = (const float*)d_in[28]; W.sa_b  = (const float*)d_in[29];
  W.sa_ow = (const float*)d_in[30]; W.sa_ob = (const float*)d_in[31];
  W.ca_w  = (const float*)d_in[32]; W.ca_b  = (const float*)d_in[33];
  W.ca_ow = (const float*)d_in[34]; W.ca_ob = (const float*)d_in[35];
  W.f1w   = (const float*)d_in[36]; W.f1b   = (const float*)d_in[37];
  W.f2w   = (const float*)d_in[38]; W.f2b   = (const float*)d_in[39];
  W.lng   = (const float*)d_in[40]; W.lnb   = (const float*)d_in[41];

  // workspace layout (floats). SZ = 16384*256 = 4,194,304
  float* w = (float*)d_ws;
  const long long SZ = 16384LL * 256;
  float* q      = w;
  float* kbuf   = q + SZ;
  float* vbuf   = kbuf + SZ;
  float* attn   = vbuf + SZ;
  float* proj   = attn + SZ;
  float* hbuf   = proj + SZ;
  float* act_o  = hbuf + SZ;
  float* act_c  = act_o + SZ;
  float* act_ar = act_c + SZ;
  float* prev   = act_ar + SZ;
  float* sbuf   = prev + SZ;            // 64*512*512 = 16,777,216 floats
  // encoder feature buffers alias sbuf (scores unused until first MHA)
  float* o_enc  = sbuf;                 // 16384 x 72
  float* c_enc  = o_enc + 16384LL * 72; // 16384 x 32
  float* ar_enc = c_enc + 16384LL * 32; // 16384 x 40

  // ---- encoders ----
  encode_kernel<<<64, 256, 0, stream>>>(o_points, o_numbers, pe_sin_w, pe_sin_b,
                                        pe_cos_w, pe_cos_b, o_num_w, o_num_b,
                                        o_enc, 16384, 2, 1);
  encode_kernel<<<64, 256, 0, stream>>>(c_points, nullptr, pe_sin_w, pe_sin_b,
                                        pe_cos_w, pe_cos_b, nullptr, nullptr,
                                        c_enc, 16384, 1, 0);
  encode_kernel<<<64, 256, 0, stream>>>(ar_points, ar_numbers, pe_sin_w, pe_sin_b,
                                        pe_cos_w, pe_cos_b, ar_num_w, ar_num_b,
                                        ar_enc, 16384, 1, 1);
  // order: leaky(enc@W1^T+b1)@W2^T+b2  (K=72 zero-padded inside tiles)
  gemm_nt(stream, 2, o_enc, 72, 0, 0, o_mlp1_w, 72, 0, 0, hbuf, 256, 0, 0,
          o_mlp1_b, 16384, 256, 72, 1, 1, 1.0f);
  gemm_nt(stream, 0, hbuf, 256, 0, 0, o_mlp2_w, 256, 0, 0, act_o, 256, 0, 0,
          o_mlp2_b, 16384, 256, 256, 1, 1, 1.0f);
  // courier (K=32)
  gemm_nt(stream, 2, c_enc, 32, 0, 0, c_mlp1_w, 32, 0, 0, hbuf, 256, 0, 0,
          c_mlp1_b, 16384, 256, 32, 1, 1, 1.0f);
  gemm_nt(stream, 0, hbuf, 256, 0, 0, c_mlp2_w, 256, 0, 0, act_c, 256, 0, 0,
          c_mlp2_b, 16384, 256, 256, 1, 1, 1.0f);
  // active-route (K=40)
  gemm_nt(stream, 2, ar_enc, 40, 0, 0, ar_mlp1_w, 40, 0, 0, hbuf, 256, 0, 0,
          ar_mlp1_b, 16384, 256, 40, 1, 1, 1.0f);
  gemm_nt(stream, 0, hbuf, 256, 0, 0, ar_mlp2_w, 256, 0, 0, act_ar, 256, 0, 0,
          ar_mlp2_b, 16384, 256, 256, 1, 1, 1.0f);

  // ---- encoder_AR: layers 0 (ord<-o|ar), 1 (crr<-c|ar) ----
  run_dec_layer(stream, act_o, act_ar, o_len, ar_len, 0, W, q, kbuf, vbuf, sbuf, attn, proj, hbuf);
  run_dec_layer(stream, act_c, act_ar, c_len, ar_len, 1, W, q, kbuf, vbuf, sbuf, attn, proj, hbuf);

  // ---- encoder_OC: 2 rounds of (new_ord, new_crr) ----
  for (int li = 0; li < 2; ++li) {
    hipMemcpyAsync(prev, act_o, (size_t)SZ * sizeof(float),
                   hipMemcpyDeviceToDevice, stream);
    run_dec_layer(stream, act_o, act_c, o_len, c_len, 2 + 2 * li, W, q, kbuf, vbuf, sbuf, attn, proj, hbuf);
    run_dec_layer(stream, act_c, prev,  c_len, o_len, 3 + 2 * li, W, q, kbuf, vbuf, sbuf, attn, proj, hbuf);
  }

  // ---- bipartite scores: out[b,o,c] = - ord . crr^T ----
  gemm_nt(stream, 0, act_o, 256, 512LL * 256, 0, act_c, 256, 512LL * 256, 0,
          (float*)d_out, 512, 512LL * 512, 0, nullptr, 512, 512, 256, 32, 1,
          -1.0f);
}
